// EncoderBlock_27633819583152
// MI455X (gfx1250) — matrix operand order repaired and checked by reading, not compiled, not run on hardware
//
#include <hip/hip_runtime.h>
#include <cstddef>

// ---------- types ----------
typedef __attribute__((ext_vector_type(16))) __bf16 v16bf;
typedef __attribute__((ext_vector_type(8)))  float  v8f;
typedef __attribute__((ext_vector_type(4)))  unsigned int u32x4;
typedef __attribute__((ext_vector_type(8)))  int i32x8;
typedef __attribute__((ext_vector_type(4)))  int i32x4;

struct alignas(16) U4 { unsigned x, y, z, w; };
struct alignas(16) F4 { float x, y, z, w; };

union Frag16 { U4 q[2]; v16bf v; };

__device__ __forceinline__ unsigned short f2bf(float f) {
    unsigned u = __builtin_bit_cast(unsigned, f);
    unsigned r = u + 0x7FFFu + ((u >> 16) & 1u);   // round-to-nearest-even
    return (unsigned short)(r >> 16);
}

// ---------- constants ----------
#define T_SEQ   2048
#define D_MODEL 768
#define N_HEADS 12
#define D_HEAD  64
#define D_FF    3072
#define BT      8192            // B*T = 4*2048

#define BM 256                  // block rows (8 waves x 32 rows)
#define BN 64
#define BK 32
#define AST 40                  // LDS stride (elems) for A tile rows (80B)
#define BST 40                  // LDS stride (elems) for B tile cols
#define KST 72                  // LDS stride (elems) in attention tiles (144B)

// ---------- Tensor Data Mover (CDNA5) ----------
#if defined(__has_builtin)
#if __has_builtin(__builtin_amdgcn_tensor_load_to_lds)
#define USE_TDM 1
#endif
#endif
#ifndef USE_TDM
#define USE_TDM 0
#endif

#if USE_TDM
__device__ __forceinline__ int rfl(unsigned v) {
    return __builtin_amdgcn_readfirstlane((int)v);
}
// 2D tile load, data_size = 8 bytes. Row of tile0_units*8 bytes per tensor row,
// tile1_rows rows, global row stride = stride_units*8 bytes.
// LDS padding: pad_amount DWORD-code inserted every pad_interval DWORD-code.
__device__ __forceinline__ void tdm_load_2d(unsigned lds_off, const void* gptr,
                                            unsigned tile0_units, unsigned tile1_rows,
                                            unsigned stride_units, unsigned tensor1,
                                            unsigned pad_int_code, unsigned pad_amt_code)
{
    const unsigned long long ga = (unsigned long long)(size_t)gptr;
    u32x4 g0;
    g0.x = (unsigned)rfl(1u);                                  // count=1, user mode
    g0.y = (unsigned)rfl(lds_off);                             // lds_addr
    g0.z = (unsigned)rfl((unsigned)ga);                        // global_addr[31:0]
    g0.w = (unsigned)rfl(((unsigned)(ga >> 32) & 0x01FFFFFFu) | (2u << 30)); // addr[56:32], type=2
    i32x8 g1;
    g1[0] = rfl((3u << 16) | (1u << 20) | (pad_int_code << 22) | (pad_amt_code << 25));
    g1[1] = rfl((stride_units & 0xFFFFu) << 16);               // tensor_dim0[15:0] (== stride)
    g1[2] = rfl(((stride_units >> 16) & 0xFFFFu) | ((tensor1 & 0xFFFFu) << 16));
    g1[3] = rfl(((tensor1 >> 16) & 0xFFFFu) | (tile0_units << 16)); // tile_dim0
    g1[4] = rfl(tile1_rows & 0xFFFFu);                         // tile_dim1, tile_dim2=0
    g1[5] = rfl(stride_units);                                 // tensor_dim0_stride[31:0]
    g1[6] = 0;                                                 // stride hi / dim1_stride
    g1[7] = 0;
    const i32x4 z4 = {0, 0, 0, 0};
    const i32x8 z8 = {0, 0, 0, 0, 0, 0, 0, 0};
    // 6-arg form (amdgpu-toolchain / clang-23 HIP headers)
    __builtin_amdgcn_tensor_load_to_lds(g0, g1, z4, z4, z8, 0);
}
#endif

// =====================================================================
// LayerNorm: one block per row of 768, output bf16
// =====================================================================
__global__ __launch_bounds__(256)
void ln_kernel(const float* __restrict__ x, const float* __restrict__ g,
               const float* __restrict__ bb, unsigned short* __restrict__ out)
{
    __shared__ float red[8];
    const int row = blockIdx.x, tid = threadIdx.x;
    const int wave = tid >> 5, lane = tid & 31;
    const float* xr = x + (size_t)row * D_MODEL;
    float v0 = xr[tid], v1 = xr[tid + 256], v2 = xr[tid + 512];
    float s = v0 + v1 + v2;
    #pragma unroll
    for (int m = 1; m < 32; m <<= 1) s += __shfl_xor(s, m, 32);
    if (lane == 0) red[wave] = s;
    __syncthreads();
    float tot = 0.f;
    #pragma unroll
    for (int i = 0; i < 8; ++i) tot += red[i];
    const float mu = tot * (1.f / 768.f);
    float d0 = v0 - mu, d1 = v1 - mu, d2 = v2 - mu;
    float s2 = d0 * d0 + d1 * d1 + d2 * d2;
    #pragma unroll
    for (int m = 1; m < 32; m <<= 1) s2 += __shfl_xor(s2, m, 32);
    __syncthreads();
    if (lane == 0) red[wave] = s2;
    __syncthreads();
    float tot2 = 0.f;
    #pragma unroll
    for (int i = 0; i < 8; ++i) tot2 += red[i];
    const float rs = rsqrtf(tot2 * (1.f / 768.f) + 1e-5f);
    unsigned short* orow = out + (size_t)row * D_MODEL;
    orow[tid]       = f2bf(d0 * rs * g[tid]       + bb[tid]);
    orow[tid + 256] = f2bf(d1 * rs * g[tid + 256] + bb[tid + 256]);
    orow[tid + 512] = f2bf(d2 * rs * g[tid + 512] + bb[tid + 512]);
}

// =====================================================================
// bf16 WMMA GEMM: C = A(bf16 [M,K]) * B(f32 [K,N]) + bias
// block tile 256x64, 8 waves; wave w owns rows [32w,32w+32) x 64 cols
// (2 A-frags reuse each B-frag -> 8 WMMA per 12 b128 LDS reads per K-step)
// EPI 1: gelu -> bf16 | EPI 2: + resid -> f32 | EPI 3: qkv scatter bf16
// =====================================================================
template<int EPI>
__global__ __launch_bounds__(256)
void gemm_bf16(const unsigned short* __restrict__ A,
               const float* __restrict__ Bw,
               const float* __restrict__ bias,
               const float* __restrict__ resid,
               float* __restrict__ outF,
               unsigned short* __restrict__ outH,
               unsigned short* __restrict__ outQ,
               unsigned short* __restrict__ outK,
               unsigned short* __restrict__ outV,
               int M, int N, int K)
{
    __shared__ __align__(16) unsigned short As[BM * AST];
    __shared__ __align__(16) unsigned short Bs[BN * BST];
    const int tid   = threadIdx.x;
    const int wave  = tid >> 5;
    const int lane  = tid & 31;
    const int lhalf = lane >> 4;
    const int l16   = lane & 15;
    const int m0 = blockIdx.x * BM;
    const int n0 = blockIdx.y * BN;

    v8f acc[2][4] = {};

    for (int k0 = 0; k0 < K; k0 += BK) {
        __syncthreads();
#if USE_TDM
        // ---- A tile 256x32 bf16 via Tensor Data Mover (row 64B + 16B pad) ----
        if (wave == 0) {
            tdm_load_2d((unsigned)(size_t)&As[0],
                        A + (size_t)m0 * K + k0,
                        /*tile0(8B)*/ 8, /*rows*/ BM,
                        /*stride(8B)*/ (unsigned)(K >> 2), /*tensor1*/ (unsigned)M,
                        /*pad every 16 DW*/ 3, /*pad 4 DW*/ 3);
        }
#else
        #pragma unroll
        for (int i = 0; i < 4; ++i) {
            int c = tid + i * 256;              // 1024 chunks of 8 bf16
            int row = c >> 2, cq = (c & 3) * 8;
            *reinterpret_cast<U4*>(&As[row * AST + cq]) =
                *reinterpret_cast<const U4*>(A + (size_t)(m0 + row) * K + k0 + cq);
        }
#endif
        // ---- B tile 32x64 f32 -> bf16, transposed store Bs[n][k] ----
        #pragma unroll
        for (int i = 0; i < 2; ++i) {
            int c = tid + i * 256;              // 512 chunks of 4 floats
            int kr = c >> 4, nc = (c & 15) * 4;
            F4 f = *reinterpret_cast<const F4*>(Bw + (size_t)(k0 + kr) * N + n0 + nc);
            Bs[(nc + 0) * BST + kr] = f2bf(f.x);
            Bs[(nc + 1) * BST + kr] = f2bf(f.y);
            Bs[(nc + 2) * BST + kr] = f2bf(f.z);
            Bs[(nc + 3) * BST + kr] = f2bf(f.w);
        }
#if USE_TDM
        if (wave == 0) __builtin_amdgcn_s_wait_tensorcnt(0);
#endif
        __syncthreads();
        // ---- fragments + WMMA ----
        Frag16 a0, a1;
        const int ar0   = wave * 32 + l16;
        const int abase = lhalf * 8;
        a0.q[0] = *reinterpret_cast<const U4*>(&As[ar0 * AST + abase]);
        a0.q[1] = *reinterpret_cast<const U4*>(&As[ar0 * AST + 16 + abase]);
        a1.q[0] = *reinterpret_cast<const U4*>(&As[(ar0 + 16) * AST + abase]);
        a1.q[1] = *reinterpret_cast<const U4*>(&As[(ar0 + 16) * AST + 16 + abase]);
        const int koff = lhalf * 8;
        #pragma unroll
        for (int ns = 0; ns < 4; ++ns) {
            Frag16 b;
            const int bn = ns * 16 + l16;
            b.q[0] = *reinterpret_cast<const U4*>(&Bs[bn * BST + koff]);
            b.q[1] = *reinterpret_cast<const U4*>(&Bs[bn * BST + koff + 16]);
            acc[0][ns] = __builtin_amdgcn_wmma_f32_16x16x32_bf16(
                false, a0.v, false, b.v, (short)0, acc[0][ns], false, false);
            acc[1][ns] = __builtin_amdgcn_wmma_f32_16x16x32_bf16(
                false, a1.v, false, b.v, (short)0, acc[1][ns], false, false);
        }
    }

    // ---- epilogue ----
    #pragma unroll
    for (int ns = 0; ns < 4; ++ns) {
        const int gn = n0 + ns * 16 + l16;
        const float bv = bias[gn];
        #pragma unroll
        for (int sub = 0; sub < 2; ++sub) {
            #pragma unroll
            for (int r = 0; r < 8; ++r) {
                const int gm = m0 + wave * 32 + sub * 16 + r + 8 * lhalf;
                float v = acc[sub][ns][r] + bv;
                if constexpr (EPI == 1) {           // tanh-approx GELU -> bf16
                    float c = v + 0.044715f * v * v * v;
                    v = 0.5f * v * (1.f + tanhf(0.7978845608028654f * c));
                    outH[(size_t)gm * N + gn] = f2bf(v);
                } else if constexpr (EPI == 2) {    // residual -> f32
                    outF[(size_t)gm * N + gn] = resid[(size_t)gm * N + gn] + v;
                } else if constexpr (EPI == 3) {    // QKV scatter
                    const int which = gn / D_MODEL;
                    const int d     = gn % D_MODEL;
                    const int head  = d >> 6, dd = d & 63;
                    const int bidx  = gm >> 11, t = gm & (T_SEQ - 1);
                    unsigned short* dst = (which == 0) ? outQ : (which == 1 ? outK : outV);
                    const float vv = (which == 0) ? v * 0.125f : v;  // fold 1/sqrt(64) into Q
                    dst[((size_t)(bidx * N_HEADS + head) * T_SEQ + t) * D_HEAD + dd] = f2bf(vv);
                } else {
                    outH[(size_t)gm * N + gn] = f2bf(v);
                }
            }
        }
    }
}

// =====================================================================
// Flash attention: grid (T/128, B*H), 256 threads.
// Q,K,V bf16 [B,H,T,64]; out bf16 [B,T,768]. Q pre-scaled by 1/8.
// =====================================================================
__global__ __launch_bounds__(256)
void attn_kernel(const unsigned short* __restrict__ Q,
                 const unsigned short* __restrict__ Kb,
                 const unsigned short* __restrict__ Vb,
                 unsigned short* __restrict__ Out)
{
    __shared__ __align__(16) unsigned short Ks[64 * KST];   // [kcol][d]
    __shared__ __align__(16) unsigned short Vs[64 * KST];   // [d][kcol]
    __shared__ __align__(16) unsigned short Ps[128 * KST];  // [qrow][kcol]
    const int tid   = threadIdx.x;
    const int wave  = tid >> 5;
    const int lane  = tid & 31;
    const int lhalf = lane >> 4;
    const int l16   = lane & 15;
    const int bh = blockIdx.y;                  // b*12 + h
    const int q0 = blockIdx.x * 128;
    const unsigned short* qp = Q  + (size_t)bh * T_SEQ * D_HEAD;
    const unsigned short* kp = Kb + (size_t)bh * T_SEQ * D_HEAD;
    const unsigned short* vp = Vb + (size_t)bh * T_SEQ * D_HEAD;

    // Q fragments for this wave's 16 rows (dh=64 -> 2 k-steps), straight from HBM
    Frag16 qf[2];
    {
        const unsigned short* qr = qp + (size_t)(q0 + wave * 16 + l16) * D_HEAD;
        const int base = lhalf * 8;
        #pragma unroll
        for (int s = 0; s < 2; ++s) {
            qf[s].q[0] = *reinterpret_cast<const U4*>(qr + s * 32 + base);
            qf[s].q[1] = *reinterpret_cast<const U4*>(qr + s * 32 + 16 + base);
        }
    }

    float m[8], l[8];
    v8f O[4] = {};
    #pragma unroll
    for (int r = 0; r < 8; ++r) { m[r] = -1e30f; l[r] = 0.f; }

    const int koff = lhalf * 8;
    for (int kc = 0; kc < T_SEQ; kc += 64) {
        __syncthreads();
#if USE_TDM
        // ---- K chunk [64][64]bf16 via TDM (row 128B + 16B pad -> stride 144B) ----
        if (wave == 0) {
            tdm_load_2d((unsigned)(size_t)&Ks[0], kp + (size_t)kc * D_HEAD,
                        /*tile0(8B)*/ 16, /*rows*/ 64,
                        /*stride(8B)*/ 16, /*tensor1*/ T_SEQ,
                        /*pad every 32 DW*/ 4, /*pad 4 DW*/ 3);
        }
#else
        #pragma unroll
        for (int i = 0; i < 4; ++i) {
            int c = tid + i * 256;              // 1024 chunks of 8 bf16
            int kr = c >> 3, d8 = (c & 7) * 8;
            *reinterpret_cast<U4*>(&Ks[kr * KST + d8]) =
                *reinterpret_cast<const U4*>(kp + (size_t)(kc + kr) * D_HEAD + d8);
        }
#endif
        // ---- V chunk transposed: Vs[d*KST + kcol] (manual: needs transpose) ----
        #pragma unroll
        for (int i = 0; i < 4; ++i) {
            int c = tid + i * 256;
            int kr = c >> 3, d8 = (c & 7) * 8;
            U4 u = *reinterpret_cast<const U4*>(vp + (size_t)(kc + kr) * D_HEAD + d8);
            unsigned wv[4] = { u.x, u.y, u.z, u.w };
            #pragma unroll
            for (int j = 0; j < 4; ++j) {
                Vs[(d8 + 2 * j + 0) * KST + kr] = (unsigned short)(wv[j] & 0xFFFFu);
                Vs[(d8 + 2 * j + 1) * KST + kr] = (unsigned short)(wv[j] >> 16);
            }
        }
#if USE_TDM
        if (wave == 0) __builtin_amdgcn_s_wait_tensorcnt(0);
#endif
        __syncthreads();

        // ---- S = Q @ K^T (per wave: 16 q-rows x 64 k-cols) ----
        v8f S[4] = {};
        #pragma unroll
        for (int ns = 0; ns < 4; ++ns) {
            const int n = ns * 16 + l16;
            #pragma unroll
            for (int s = 0; s < 2; ++s) {
                Frag16 b;
                b.q[0] = *reinterpret_cast<const U4*>(&Ks[n * KST + s * 32 + koff]);
                b.q[1] = *reinterpret_cast<const U4*>(&Ks[n * KST + s * 32 + koff + 16]);
                S[ns] = __builtin_amdgcn_wmma_f32_16x16x32_bf16(
                    false, qf[s].v, false, b.v, (short)0, S[ns], false, false);
            }
        }

        // ---- online softmax ----
        float rsum[8];
        #pragma unroll
        for (int r = 0; r < 8; ++r) {
            float v = fmaxf(fmaxf(S[0][r], S[1][r]), fmaxf(S[2][r], S[3][r]));
            #pragma unroll
            for (int msk = 1; msk < 16; msk <<= 1)
                v = fmaxf(v, __shfl_xor(v, msk, 32));
            const float mn = fmaxf(m[r], v);
            const float alpha = __expf(m[r] - mn);
            m[r] = mn;
            l[r] *= alpha;
            #pragma unroll
            for (int ns = 0; ns < 4; ++ns) O[ns][r] *= alpha;
            rsum[r] = 0.f;
        }
        const int prow0 = wave * 16 + 8 * lhalf;
        #pragma unroll
        for (int ns = 0; ns < 4; ++ns) {
            #pragma unroll
            for (int r = 0; r < 8; ++r) {
                const float p = __expf(S[ns][r] - m[r]);
                rsum[r] += p;
                Ps[(prow0 + r) * KST + ns * 16 + l16] = f2bf(p);
            }
        }
        #pragma unroll
        for (int r = 0; r < 8; ++r) {
            float v = rsum[r];
            #pragma unroll
            for (int msk = 1; msk < 16; msk <<= 1)
                v += __shfl_xor(v, msk, 32);
            l[r] += v;
        }

        // ---- O += P @ V (wave-private P region; in-wave ds ordering) ----
        const int arow  = wave * 16 + l16;
        const int abase = lhalf * 8;
        #pragma unroll
        for (int s = 0; s < 2; ++s) {
            Frag16 a;
            a.q[0] = *reinterpret_cast<const U4*>(&Ps[arow * KST + s * 32 + abase]);
            a.q[1] = *reinterpret_cast<const U4*>(&Ps[arow * KST + s * 32 + 16 + abase]);
            #pragma unroll
            for (int ns = 0; ns < 4; ++ns) {
                Frag16 b;
                const int n = ns * 16 + l16;
                b.q[0] = *reinterpret_cast<const U4*>(&Vs[n * KST + s * 32 + koff]);
                b.q[1] = *reinterpret_cast<const U4*>(&Vs[n * KST + s * 32 + koff + 16]);
                O[ns] = __builtin_amdgcn_wmma_f32_16x16x32_bf16(
                    false, a.v, false, b.v, (short)0, O[ns], false, false);
            }
        }
    }

    // ---- write attn output in [B,T,768] bf16 ----
    const int b = bh / N_HEADS, h = bh % N_HEADS;
    #pragma unroll
    for (int ns = 0; ns < 4; ++ns) {
        const int d = ns * 16 + l16;
        #pragma unroll
        for (int r = 0; r < 8; ++r) {
            const int t = q0 + wave * 16 + r + 8 * lhalf;
            Out[((size_t)(b * T_SEQ + t)) * D_MODEL + h * D_HEAD + d] =
                f2bf(O[ns][r] / l[r]);
        }
    }
}

// =====================================================================
// launcher
// =====================================================================
extern "C" void kernel_launch(void* const* d_in, const int* in_sizes, int n_in,
                              void* d_out, int out_size, void* d_ws, size_t ws_size,
                              hipStream_t stream) {
    (void)in_sizes; (void)n_in; (void)out_size; (void)ws_size;
    const float* x     = (const float*)d_in[0];
    const float* ln1_g = (const float*)d_in[1];
    const float* ln1_b = (const float*)d_in[2];
    const float* qkv_w = (const float*)d_in[3];
    const float* qkv_b = (const float*)d_in[4];
    const float* ao_w  = (const float*)d_in[5];
    const float* ao_b  = (const float*)d_in[6];
    const float* ln2_g = (const float*)d_in[7];
    const float* ln2_b = (const float*)d_in[8];
    const float* ff1_w = (const float*)d_in[9];
    const float* ff1_b = (const float*)d_in[10];
    const float* ff2_w = (const float*)d_in[11];
    const float* ff2_b = (const float*)d_in[12];
    float* out = (float*)d_out;

    char* ws = (char*)d_ws;
    const size_t SZ_BF = (size_t)BT * D_MODEL * 2;          // 12.6 MB
    unsigned short* qb  = (unsigned short*)ws;  ws += SZ_BF;
    unsigned short* kb  = (unsigned short*)ws;  ws += SZ_BF;
    unsigned short* vb  = (unsigned short*)ws;  ws += SZ_BF;
    unsigned short* hb  = (unsigned short*)ws;  ws += SZ_BF; // LN out (reused)
    unsigned short* ab  = (unsigned short*)ws;  ws += SZ_BF; // attn out
    float*          x2  = (float*)ws;           ws += (size_t)BT * D_MODEL * 4;
    unsigned short* ffb = (unsigned short*)ws;  ws += (size_t)BT * D_FF * 2;

    // 1) h = LN1(x)
    ln_kernel<<<BT, 256, 0, stream>>>(x, ln1_g, ln1_b, hb);
    // 2) qkv = h @ qkv_w + b -> scattered q(*0.125)/k/v
    gemm_bf16<3><<<dim3(BT / BM, (3 * D_MODEL) / BN), 256, 0, stream>>>(
        hb, qkv_w, qkv_b, nullptr, nullptr, nullptr, qb, kb, vb,
        BT, 3 * D_MODEL, D_MODEL);
    // 3) attention
    attn_kernel<<<dim3(T_SEQ / 128, 4 * N_HEADS), 256, 0, stream>>>(qb, kb, vb, ab);
    // 4) x2 = x + attn @ ao_w + b
    gemm_bf16<2><<<dim3(BT / BM, D_MODEL / BN), 256, 0, stream>>>(
        ab, ao_w, ao_b, x, x2, nullptr, nullptr, nullptr, nullptr,
        BT, D_MODEL, D_MODEL);
    // 5) h = LN2(x2)
    ln_kernel<<<BT, 256, 0, stream>>>(x2, ln2_g, ln2_b, hb);
    // 6) ff = gelu(h @ ff1_w + b)
    gemm_bf16<1><<<dim3(BT / BM, D_FF / BN), 256, 0, stream>>>(
        hb, ff1_w, ff1_b, nullptr, nullptr, ffb, nullptr, nullptr, nullptr,
        BT, D_FF, D_MODEL);
    // 7) out = x2 + ff @ ff2_w + b
    gemm_bf16<2><<<dim3(BT / BM, D_MODEL / BN), 256, 0, stream>>>(
        ffb, ff2_w, ff2_b, x2, out, nullptr, nullptr, nullptr, nullptr,
        BT, D_MODEL, D_FF);
}
